// AttentionLayer_75179107549656
// MI455X (gfx1250) — compile-verified, matrix operand order checked by reading, not run on hardware
//
#include <hip/hip_runtime.h>

typedef __attribute__((ext_vector_type(16))) _Float16 v16h;
typedef __attribute__((ext_vector_type(8)))  _Float16 v8h;
typedef __attribute__((ext_vector_type(8)))  float    v8f;

#define Bsz  2
#define Sseq 2048
#define Dmod 1024
#define Hn   16
#define HDim 64
#define Mrows (Bsz * Sseq)   // 4096

// ---- CDNA5 async copy (global -> LDS, 16B per lane, ASYNCcnt-tracked) ------
static __device__ __forceinline__ void async_copy_b128(const _Float16* gsrc,
                                                       unsigned lds_off) {
    asm volatile("global_load_async_to_lds_b128 %0, %1, off"
                 :
                 : "v"(lds_off), "v"((unsigned long long)(uintptr_t)gsrc)
                 : "memory");
}
static __device__ __forceinline__ void wait_async0() {
    asm volatile("s_wait_asynccnt 0" ::: "memory");
}
static __device__ __forceinline__ unsigned lds_addr(const void* p) {
    return (unsigned)(uintptr_t)p;  // low 32 bits of generic LDS ptr = DS offset
}

// ---- fragment helper -------------------------------------------------------
// 16-bit A/B fragment (v16h = 8 VGPRs): elements 0..7 = K {0..7} (+lane-half
// offset 8 for lanes 16..31), elements 8..15 = K {16..23} (+lane-half offset).
static __device__ __forceinline__ v16h load_frag(const _Float16* p) {
    v8h lo = *(const v8h*)(p);
    v8h hi = *(const v8h*)(p + 16);
    v16h r;
#pragma unroll
    for (int i = 0; i < 8; i++) { r[i] = lo[i]; r[i + 8] = hi[i]; }
    return r;
}

static __device__ __forceinline__ v8f wmma_f16(v16h a, v16h b, v8f c) {
    return __builtin_amdgcn_wmma_f32_16x16x32_f16(false, a, false, b, (short)0,
                                                  c, false, false);
}

// ---- fp32 -> fp16 conversion kernels --------------------------------------
__global__ void cvt_f32_f16(const float* __restrict__ in,
                            _Float16* __restrict__ out, int n) {
    int i = blockIdx.x * blockDim.x + threadIdx.x;
    int stride = gridDim.x * blockDim.x;
    for (; i < n; i += stride) out[i] = (_Float16)in[i];
}

// W[K][N] fp32 -> Wt[N][K] fp16 (transposed for contiguous-K B fragments)
__global__ void cvt_transpose(const float* __restrict__ W,
                              _Float16* __restrict__ Wt) {
    int i = blockIdx.x * blockDim.x + threadIdx.x;
    int stride = gridDim.x * blockDim.x;
    for (; i < Dmod * Dmod; i += stride) {
        int k = i / Dmod, n = i % Dmod;
        Wt[(size_t)n * Dmod + k] = (_Float16)W[i];
    }
}

// ---- WMMA GEMM: C[M,N] = A[M,K] * Bt[N,K]^T + bias -------------------------
// 8 waves/block share one 64-column B panel, staged to LDS with async copies
// (double buffered). Each wave owns a 32x64 output tile (2x B-fragment reuse).
// MODE 0: f16 head-split [B,H,S,HD] | MODE 1: f16 [B,H,HD,S] | MODE 2: f32 [M,N]
template <int MODE>
__global__ __launch_bounds__(256) void gemm_wmma(
    const _Float16* __restrict__ A, const _Float16* __restrict__ Bt,
    const float* __restrict__ bias, void* __restrict__ out) {
    constexpr int K = Dmod, N = Dmod;
    __shared__ _Float16 ldsB[2][64 * 32];  // [col 0..63][k 0..31], 4KB each

    const int tid  = threadIdx.x;
    const int lane = tid & 31;
    const int wid  = tid >> 5;
    const int bn   = blockIdx.x & 15;                  // 16 column panels of 64
    const int bm   = blockIdx.x >> 4;                  // 16 row groups of 256
    const int m0 = (bm * 8 + wid) * 32;                // 32 rows per wave
    const int n0 = bn * 64;
    const int l15 = lane & 15;
    const int hiH = (lane >= 16);
    const int rbase = hiH ? 8 : 0;

    // cooperative staging: thread -> (column, 16-byte segment of its 32 k's)
    const int scol = tid >> 2;        // 0..63
    const int sseg = (tid & 3) * 8;   // f16 elements 0,8,16,24
    const _Float16* gB = Bt + (size_t)(n0 + scol) * K + sseg;
    const unsigned ldsB_off[2] = {
        lds_addr(&ldsB[0][scol * 32 + sseg]),
        lds_addr(&ldsB[1][scol * 32 + sseg])};

    async_copy_b128(gB, ldsB_off[0]);
    wait_async0();
    __syncthreads();

    const _Float16* arow0 = A + (size_t)(m0 + l15) * K + (hiH ? 8 : 0);
    const _Float16* arow1 = arow0 + (size_t)16 * K;

    v8f acc[2][4] = {};
    int cur = 0;
#pragma unroll 1
    for (int kc = 0; kc < K; kc += 32) {
        if (kc + 32 < K)
            async_copy_b128(gB + kc + 32, ldsB_off[cur ^ 1]);
        __builtin_prefetch(arow0 + kc + 128, 0, 0);
        __builtin_prefetch(arow1 + kc + 128, 0, 0);
        // hoist all fragment loads so DS/VMEM latency overlaps the WMMA chain
        v16h a0 = load_frag(arow0 + kc);
        v16h a1 = load_frag(arow1 + kc);
        const _Float16* bpan = &ldsB[cur][0];
        v16h bfr[4];
#pragma unroll
        for (int t = 0; t < 4; t++)
            bfr[t] = load_frag(bpan + (t * 16 + l15) * 32 + (hiH ? 8 : 0));
#pragma unroll
        for (int t = 0; t < 4; t++) acc[0][t] = wmma_f16(a0, bfr[t], acc[0][t]);
#pragma unroll
        for (int t = 0; t < 4; t++) acc[1][t] = wmma_f16(a1, bfr[t], acc[1][t]);
        wait_async0();
        __syncthreads();
        cur ^= 1;
    }

    // C layout: lane holds column n0+t*16+l15; VGPR i holds row rbase+i.
#pragma unroll
    for (int r = 0; r < 2; r++) {
#pragma unroll
        for (int t = 0; t < 4; t++) {
            int c = n0 + t * 16 + l15;
            float bsv = bias[c];
#pragma unroll
            for (int i = 0; i < 8; i++) {
                int gr = m0 + r * 16 + rbase + i;
                float v = acc[r][t][i] + bsv;
                if (MODE == 2) {
                    ((float*)out)[(size_t)gr * N + c] = v;
                } else {
                    int b = gr / Sseq, s = gr % Sseq;
                    int h = c / HDim, hd = c % HDim;
                    _Float16* o = (_Float16*)out;
                    if (MODE == 0)
                        o[((size_t)(b * Hn + h) * Sseq + s) * HDim + hd] =
                            (_Float16)v;
                    else
                        o[((size_t)(b * Hn + h) * HDim + hd) * Sseq + s] =
                            (_Float16)v;
                }
            }
        }
    }
}

// ---- flash attention: 8 waves/block share (b,h); one wave per query tile ---
__global__ __launch_bounds__(256) void attn_wmma(
    const _Float16* __restrict__ Qh,   // [B,H,S,HD] f16
    const _Float16* __restrict__ Kh,   // [B,H,S,HD] f16
    const _Float16* __restrict__ Vt,   // [B,H,HD,S] f16
    _Float16* __restrict__ att) {      // [B,S,D]    f16
    __shared__ _Float16 kbuf[2][32 * HDim];  // [key 0..31][hd 0..63], 4KB
    __shared__ _Float16 vbuf[2][HDim * 32];  // [hd 0..63][key 0..31], 4KB
    __shared__ float lds_sc[8][16 * 32];     // per-wave probability tile
    __shared__ float lds_f[8][16];           // per-wave row factors

    const int tid  = threadIdx.x;
    const int lane = tid & 31;
    const int wid  = tid >> 5;
    const int gw   = blockIdx.x * 8 + wid;
    const int qt   = gw & 127;            // 128 query tiles per (b,h)
    const int bh   = gw >> 7;             // same for all 8 waves in a block
    const int q0   = qt * 16;
    const int l15  = lane & 15;
    const int hiH  = (lane >= 16);
    const int rbase = hiH ? 8 : 0;
    const float scale = 0.125f;           // 1/sqrt(64)

    const _Float16* kpan = Kh + (size_t)bh * Sseq * HDim;
    const _Float16* vpan = Vt + (size_t)bh * HDim * Sseq;

    // staging addresses: K chunk is 4KB contiguous; V chunk is 64 rows x 64B
    const _Float16* gK = kpan + tid * 8;                       // + kb*HDim
    const int vrow = tid >> 2;
    const int vseg = (tid & 3) * 8;
    const _Float16* gV = vpan + (size_t)vrow * Sseq + vseg;    // + kb
    const unsigned koff[2] = {lds_addr(&kbuf[0][tid * 8]),
                              lds_addr(&kbuf[1][tid * 8])};
    const unsigned voff[2] = {lds_addr(&vbuf[0][vrow * 32 + vseg]),
                              lds_addr(&vbuf[1][vrow * 32 + vseg])};

    // Q A-fragments: 16 queries x 64 head-dims, as two K=32 chunks
    const _Float16* qbase =
        Qh + ((size_t)bh * Sseq + q0 + l15) * HDim + (hiH ? 8 : 0);
    v16h qa0 = load_frag(qbase);
    v16h qa1 = load_frag(qbase + 32);

    async_copy_b128(gK, koff[0]);
    async_copy_b128(gV, voff[0]);
    wait_async0();
    __syncthreads();

    float row_m = -1e30f, row_sum = 0.0f;  // live in lanes 0..15
    v8f acc[4] = {};
    float* sc_l = lds_sc[wid];
    float* f_l  = lds_f[wid];

    int cur = 0;
#pragma unroll 1
    for (int kb = 0; kb < Sseq; kb += 32) {
        if (kb + 32 < Sseq) {
            async_copy_b128(gK + (size_t)(kb + 32) * HDim, koff[cur ^ 1]);
            async_copy_b128(gV + kb + 32, voff[cur ^ 1]);
        }
        // ---- scores: Q(16x64) . K^T(64x32); hoist all 4 K fragments first
        const _Float16* kb_l = &kbuf[cur][0];
        const _Float16* kp0 = kb_l + l15 * HDim + (hiH ? 8 : 0);
        const _Float16* kp1 = kp0 + 16 * HDim;
        v16h kb00 = load_frag(kp0);
        v16h kb01 = load_frag(kp0 + 32);
        v16h kb10 = load_frag(kp1);
        v16h kb11 = load_frag(kp1 + 32);
        v8f sc0 = {}, sc1 = {};
        sc0 = wmma_f16(qa0, kb00, sc0);
        sc1 = wmma_f16(qa0, kb10, sc1);
        sc0 = wmma_f16(qa1, kb01, sc0);
        sc1 = wmma_f16(qa1, kb11, sc1);
#pragma unroll
        for (int i = 0; i < 8; i++) {
            sc_l[(rbase + i) * 32 + l15]      = sc0[i] * scale;
            sc_l[(rbase + i) * 32 + 16 + l15] = sc1[i] * scale;
        }
        __syncthreads();

        // ---- online softmax row stats (lane = query row)
        if (lane < 16) {
            float* row = sc_l + lane * 32;
            float mx = -1e30f;
#pragma unroll
            for (int j = 0; j < 32; j++) mx = fmaxf(mx, row[j]);
            float nm = fmaxf(row_m, mx);
            float f = __expf(row_m - nm);
            float s = 0.f;
#pragma unroll
            for (int j = 0; j < 32; j++) {
                float p = __expf(row[j] - nm);
                row[j] = p;
                s += p;
            }
            row_sum = row_sum * f + s;
            row_m = nm;
            f_l[lane] = f;
        }
        __syncthreads();

        // ---- rescale accumulators by exp(m_old - m_new)
#pragma unroll
        for (int t = 0; t < 4; t++)
#pragma unroll
            for (int i = 0; i < 8; i++) acc[t][i] *= f_l[rbase + i];

        // ---- P A-fragment from LDS (transpose C-layout -> A-layout)
        const float* prow = sc_l + l15 * 32;
        int cb = hiH ? 8 : 0;
        v16h pa;
#pragma unroll
        for (int j = 0; j < 8; j++) {
            pa[j]     = (_Float16)prow[cb + j];
            pa[8 + j] = (_Float16)prow[cb + 16 + j];
        }

        // ---- PV: P(16x32) . V(32x64); hoist all 4 V fragments first
        const _Float16* vb_l = &vbuf[cur][0];
        v16h vfr[4];
#pragma unroll
        for (int t = 0; t < 4; t++)
            vfr[t] = load_frag(vb_l + (t * 16 + l15) * 32 + (hiH ? 8 : 0));
#pragma unroll
        for (int t = 0; t < 4; t++) acc[t] = wmma_f16(pa, vfr[t], acc[t]);

        wait_async0();
        __syncthreads();  // next chunk staged; LDS tiles free for reuse
        cur ^= 1;
    }

    if (lane < 16) f_l[lane] = 1.0f / row_sum;
    __syncthreads();

    const int b = bh / Hn, h = bh % Hn;
#pragma unroll
    for (int t = 0; t < 4; t++) {
        int hd = t * 16 + l15;
#pragma unroll
        for (int i = 0; i < 8; i++) {
            int s = q0 + rbase + i;
            att[((size_t)(b * Sseq + s)) * Dmod + h * HDim + hd] =
                (_Float16)(acc[t][i] * f_l[rbase + i]);
        }
    }
}

// ---- launch ----------------------------------------------------------------
extern "C" void kernel_launch(void* const* d_in, const int* in_sizes, int n_in,
                              void* d_out, int out_size, void* d_ws,
                              size_t ws_size, hipStream_t stream) {
    const float* x  = (const float*)d_in[0];
    const float* Wq = (const float*)d_in[1];
    const float* bq = (const float*)d_in[2];
    const float* Wk = (const float*)d_in[3];
    const float* bk = (const float*)d_in[4];
    const float* Wv = (const float*)d_in[5];
    const float* bv = (const float*)d_in[6];
    const float* Wo = (const float*)d_in[7];
    const float* bo = (const float*)d_in[8];

    // Workspace layout (48 MB total)
    char* ws = (char*)d_ws;
    _Float16* xh  = (_Float16*)(ws);                  //  8 MB  x as f16
    _Float16* wqt = (_Float16*)(ws + (8u  << 20));    //  2 MB  Wq^T f16
    _Float16* wkt = (_Float16*)(ws + (10u << 20));    //  2 MB
    _Float16* wvt = (_Float16*)(ws + (12u << 20));    //  2 MB
    _Float16* wot = (_Float16*)(ws + (14u << 20));    //  2 MB
    _Float16* Qh  = (_Float16*)(ws + (16u << 20));    //  8 MB  [B,H,S,HD]
    _Float16* Kh  = (_Float16*)(ws + (24u << 20));    //  8 MB  [B,H,S,HD]
    _Float16* Vt  = (_Float16*)(ws + (32u << 20));    //  8 MB  [B,H,HD,S]
    _Float16* att = (_Float16*)(ws + (40u << 20));    //  8 MB  [B,S,D]

    cvt_f32_f16<<<2048, 256, 0, stream>>>(x, xh, Mrows * Dmod);
    cvt_transpose<<<1024, 256, 0, stream>>>(Wq, wqt);
    cvt_transpose<<<1024, 256, 0, stream>>>(Wk, wkt);
    cvt_transpose<<<1024, 256, 0, stream>>>(Wv, wvt);
    cvt_transpose<<<1024, 256, 0, stream>>>(Wo, wot);

    gemm_wmma<0><<<256, 256, 0, stream>>>(xh, wqt, bq, (void*)Qh);
    gemm_wmma<0><<<256, 256, 0, stream>>>(xh, wkt, bk, (void*)Kh);
    gemm_wmma<1><<<256, 256, 0, stream>>>(xh, wvt, bv, (void*)Vt);

    attn_wmma<<<512, 256, 0, stream>>>(Qh, Kh, Vt, att);

    gemm_wmma<2><<<256, 256, 0, stream>>>(att, wot, bo, d_out);
}
